// GRU_19971597927128
// MI455X (gfx1250) — matrix operand order repaired and checked by reading, not compiled, not run on hardware
//
#include <hip/hip_runtime.h>

// ---------------------------------------------------------------------------
// GRU (B=64,T=256,H=1024) + fc1(leaky_relu) + fc2, bf16 WMMA on gfx1250.
//   - cast f32->bf16 kernels
//   - persistent GRU recurrence: fused x/h projections in ONE k-loop,
//     h in registers, per-block weight rows staged once into LDS with a
//     column-major padded layout so all DS reads use immediate offsets
//   - fc1/fc2 one-tile-per-wave WMMA GEMMs
// ---------------------------------------------------------------------------

static constexpr int Bd = 64;    // batch
static constexpr int Td = 256;   // seq len
static constexpr int Hd = 1024;  // hidden
static constexpr int Ed = 1024;  // output embed
static constexpr int NWG = 64;   // workgroups in persistent recurrence
static constexpr int LROW = Hd + 8;            // padded row (2064 B)
static constexpr int COLSTR = 6 * LROW + 88;   // 6280 elem = 12560 B; %256==16
                                               // -> lane i hits banks 4i..4i+3

typedef __attribute__((ext_vector_type(16))) __bf16 v16bf;
typedef __attribute__((ext_vector_type(8)))  __bf16 v8bf;
typedef __attribute__((ext_vector_type(8)))  float  v8f;

union Frag16 {
  v16bf v;
  v8bf  h[2];
};

__device__ __forceinline__ unsigned short f32_to_bf16_bits(float f) {
  union { float f; unsigned u; } v; v.f = f;
  unsigned r = v.u + 0x7FFFu + ((v.u >> 16) & 1u);  // round-to-nearest-even
  return (unsigned short)(r >> 16);
}

__device__ __forceinline__ float sigmoid_f(float x) {
  return 1.0f / (1.0f + __expf(-x));
}

__device__ __forceinline__ float tanh_f(float x) {
  float ax = fabsf(x);
  float e  = __expf(2.0f * ax);
  float t  = 1.0f - 2.0f / (e + 1.0f);
  return copysignf(t, x);
}

// ------------------------------ cast kernel --------------------------------
__global__ void cast_f32_to_bf16(const float* __restrict__ in,
                                 unsigned short* __restrict__ out, int n) {
  for (int i = blockIdx.x * blockDim.x + threadIdx.x; i < n;
       i += gridDim.x * blockDim.x)
    out[i] = f32_to_bf16_bits(in[i]);
}

__global__ void init_barrier(unsigned* bar) { bar[0] = 0u; bar[1] = 0u; }

// ------------------------- grid-wide barrier -------------------------------
__device__ __forceinline__ void grid_barrier(unsigned* cnt, unsigned* gen,
                                             unsigned nwg) {
  __syncthreads();
  if (threadIdx.x == 0) {
    __threadfence();
    unsigned old_gen =
        __hip_atomic_load(gen, __ATOMIC_RELAXED, __HIP_MEMORY_SCOPE_AGENT);
    unsigned arrived = __hip_atomic_fetch_add(cnt, 1u, __ATOMIC_ACQ_REL,
                                              __HIP_MEMORY_SCOPE_AGENT);
    if (arrived == nwg - 1u) {
      __hip_atomic_store(cnt, 0u, __ATOMIC_RELAXED, __HIP_MEMORY_SCOPE_AGENT);
      __hip_atomic_fetch_add(gen, 1u, __ATOMIC_ACQ_REL,
                             __HIP_MEMORY_SCOPE_AGENT);
    } else {
      while (__hip_atomic_load(gen, __ATOMIC_ACQUIRE,
                               __HIP_MEMORY_SCOPE_AGENT) == old_gen)
        __builtin_amdgcn_s_sleep(2);
    }
    __threadfence();
  }
  __syncthreads();
}

// --------------------- persistent GRU recurrence ---------------------------
// 64 blocks x 128 threads. Block owns 16 output columns; wave w owns batch
// rows 16w..16w+15. The block's 96 weight rows are staged into LDS once in a
// [col][gate][row] layout (per-lane window < 11 KB => DS immediate offsets).
// One fused k-loop accumulates x_t@w_ih^T and h@w_hh^T together (r/z summed,
// h-side n separate); gate math runs on accumulator registers; h stays in
// registers across all 256 steps.
__global__ void __launch_bounds__(128) gru_recurrence(
    const unsigned short* __restrict__ x_bf,   // [B,T,H] bf16
    const unsigned short* __restrict__ w_ih,   // [3H,H] bf16
    const unsigned short* __restrict__ w_hh,   // [3H,H] bf16
    const float* __restrict__ b_ih,            // [3H]
    const float* __restrict__ b_hh,            // [3H]
    unsigned short* __restrict__ h_buf0,       // [B,H] bf16 (ping)
    unsigned short* __restrict__ h_buf1,       // [B,H] bf16 (pong)
    unsigned* __restrict__ bar) {
  __shared__ __align__(16) unsigned short wlds[16 * COLSTR];  // ~201 KB

  const int tid  = threadIdx.x;
  const int lane = tid & 31;
  const int wave = tid >> 5;
  const int l15  = lane & 15;
  const int hi   = lane >> 4;
  const int m0   = wave * 16;
  const int n    = blockIdx.x * 16 + l15;  // output column this lane owns
  const int aoff = hi * 16;                // A frag: run of 16 at k+aoff
  const int boff = hi * 16;                // B frag: run of 16 at k+boff

  // ---- stage this block's 96 weight rows into LDS (row = 128 uint4) ----
#pragma unroll 4
  for (int r = 0; r < 96; ++r) {
    const int g = r >> 4;   // 0..2 = w_ih gates, 3..5 = w_hh gates
    const int c = r & 15;
    const unsigned short* src =
        (g < 3 ? w_ih : w_hh) +
        (size_t)((g % 3) * Hd + blockIdx.x * 16 + c) * Hd;
    const uint4* s4 = (const uint4*)src;
    uint4* d4 = (uint4*)&wlds[c * COLSTR + g * LROW];
    d4[tid] = s4[tid];  // 128 threads x 16B = 2048B = one full row
  }

  const float br  = b_ih[n]          + b_hh[n];
  const float bz  = b_ih[Hd + n]     + b_hh[Hd + n];
  const float bnx = b_ih[2 * Hd + n];
  const float bnh = b_hh[2 * Hd + n];

  // per-lane LDS base: all 6 gate rows within [base, base+10.4KB)
  const unsigned short* lbase = &wlds[l15 * COLSTR + boff];

  // h0 = zeros: fill the tile this wave owns in the t=0 read buffer.
#pragma unroll
  for (int r = 0; r < 8; ++r) {
    int m = m0 + r + hi * 8;
    h_buf0[(size_t)m * Hd + n] = 0;
  }
  v8f h_reg = {};  // h[m, n] carried in registers (exactly this wave's tile)

  __syncthreads();
  grid_barrier(bar, bar + 1, NWG);

  for (int t = 0; t < Td; ++t) {
    const unsigned short* hrd = (t & 1) ? h_buf1 : h_buf0;
    unsigned short*       hwr = (t & 1) ? h_buf0 : h_buf1;

    v8f accr = {}, accz = {}, accnx = {}, accnh = {};

    const unsigned short* xrow =
        x_bf + ((size_t)(m0 + l15) * Td + t) * Hd + aoff;
    const unsigned short* hrow = hrd + (size_t)(m0 + l15) * Hd + aoff;

    // Fused k-loop: x_t @ w_ih^T and h @ w_hh^T in the same iteration.
#pragma unroll 2
    for (int ks = 0; ks < Hd / 32; ++ks) {
      const int kb = ks * 32;
      Frag16 ax, ah, ri, zi, ni, rh, zh, nh;

      const v8bf* xp = (const v8bf*)(xrow + kb);
      ax.h[0] = xp[0];
      ax.h[1] = xp[1];
      const v8bf* hp = (const v8bf*)(hrow + kb);
      ah.h[0] = hp[0];
      ah.h[1] = hp[1];

      const unsigned short* lk = lbase + kb;
      ri.h[0] = *(const v8bf*)(lk + 0 * LROW);
      ri.h[1] = *(const v8bf*)(lk + 0 * LROW + 8);
      zi.h[0] = *(const v8bf*)(lk + 1 * LROW);
      zi.h[1] = *(const v8bf*)(lk + 1 * LROW + 8);
      ni.h[0] = *(const v8bf*)(lk + 2 * LROW);
      ni.h[1] = *(const v8bf*)(lk + 2 * LROW + 8);
      rh.h[0] = *(const v8bf*)(lk + 3 * LROW);
      rh.h[1] = *(const v8bf*)(lk + 3 * LROW + 8);
      zh.h[0] = *(const v8bf*)(lk + 4 * LROW);
      zh.h[1] = *(const v8bf*)(lk + 4 * LROW + 8);
      nh.h[0] = *(const v8bf*)(lk + 5 * LROW);
      nh.h[1] = *(const v8bf*)(lk + 5 * LROW + 8);

      // interleave so same-accumulator WMMAs are 3 apart
      accr  = __builtin_amdgcn_wmma_f32_16x16x32_bf16(false, ax.v, false, ri.v,
                                                      (short)0, accr, false, false);
      accz  = __builtin_amdgcn_wmma_f32_16x16x32_bf16(false, ax.v, false, zi.v,
                                                      (short)0, accz, false, false);
      accnx = __builtin_amdgcn_wmma_f32_16x16x32_bf16(false, ax.v, false, ni.v,
                                                      (short)0, accnx, false, false);
      accr  = __builtin_amdgcn_wmma_f32_16x16x32_bf16(false, ah.v, false, rh.v,
                                                      (short)0, accr, false, false);
      accz  = __builtin_amdgcn_wmma_f32_16x16x32_bf16(false, ah.v, false, zh.v,
                                                      (short)0, accz, false, false);
      accnh = __builtin_amdgcn_wmma_f32_16x16x32_bf16(false, ah.v, false, nh.v,
                                                      (short)0, accnh, false, false);
    }

    // Gate math directly on accumulator registers (D-layout: VGPR r holds
    // m = m0 + r + 8*hi, column n fixed per lane).
#pragma unroll
    for (int r = 0; r < 8; ++r) {
      int m = m0 + r + hi * 8;
      float rg = sigmoid_f(accr[r] + br);
      float zg = sigmoid_f(accz[r] + bz);
      float ng = tanh_f(accnx[r] + bnx + rg * (accnh[r] + bnh));
      float hn = (1.0f - zg) * ng + zg * h_reg[r];
      h_reg[r] = hn;
      hwr[(size_t)m * Hd + n] = f32_to_bf16_bits(hn);
    }

    grid_barrier(bar, bar + 1, NWG);
  }
}

// ------------------------------ FC GEMMs -----------------------------------
// One 16x16 output tile per wave. M=64 fixed, N/K = 1024.
template <int LEAKY>
__global__ void __launch_bounds__(256) fc_gemm(
    const unsigned short* __restrict__ A,  // [64,K] bf16
    const unsigned short* __restrict__ W,  // [N,K] bf16
    const float* __restrict__ bias,        // [N]
    unsigned short* __restrict__ out_bf,   // used if LEAKY
    float* __restrict__ out_f32,           // used if !LEAKY
    int K, int N) {
  const int lane = threadIdx.x & 31;
  const int wave = threadIdx.x >> 5;
  const int l15  = lane & 15;
  const int hi   = lane >> 4;
  const int tile = blockIdx.x * 8 + wave;
  const int m0   = (tile & 3) * 16;
  const int n    = (tile >> 2) * 16 + l15;
  const int aoff = hi * 16;
  const int boff = hi * 16;

  const unsigned short* arow = A + (size_t)(m0 + l15) * K;
  const unsigned short* wrow = W + (size_t)n * K;

  v8f acc = {};
  for (int ks = 0; ks < K / 32; ++ks) {
    const int kb = ks * 32;
    Frag16 a, b;
    const v8bf* ap = (const v8bf*)(arow + kb + aoff);
    a.h[0] = ap[0];
    a.h[1] = ap[1];
    const v8bf* bp = (const v8bf*)(wrow + kb + boff);
    b.h[0] = bp[0];
    b.h[1] = bp[1];
    acc = __builtin_amdgcn_wmma_f32_16x16x32_bf16(false, a.v, false, b.v,
                                                  (short)0, acc, false, false);
  }
  const float bs = bias[n];
#pragma unroll
  for (int r = 0; r < 8; ++r) {
    int m = m0 + r + hi * 8;
    float y = acc[r] + bs;
    if (LEAKY) {
      y = fmaxf(y, 0.2f * y);  // leaky_relu slope 0.2
      out_bf[(size_t)m * N + n] = f32_to_bf16_bits(y);
    } else {
      out_f32[(size_t)m * N + n] = y;
    }
  }
}

// ------------------------------ launcher -----------------------------------
extern "C" void kernel_launch(void* const* d_in, const int* in_sizes, int n_in,
                              void* d_out, int out_size, void* d_ws,
                              size_t ws_size, hipStream_t stream) {
  const float* x    = (const float*)d_in[0];
  const float* w_ih = (const float*)d_in[1];
  const float* w_hh = (const float*)d_in[2];
  const float* b_ih = (const float*)d_in[3];
  const float* b_hh = (const float*)d_in[4];
  const float* w1   = (const float*)d_in[5];
  const float* b1   = (const float*)d_in[6];
  const float* w2   = (const float*)d_in[7];
  const float* b2   = (const float*)d_in[8];
  float* out = (float*)d_out;

  char* ws = (char*)d_ws;
  size_t off = 0;
  unsigned short* x_bf   = (unsigned short*)(ws + off); off += (size_t)Bd * Td * Hd * 2;
  unsigned short* wih_bf = (unsigned short*)(ws + off); off += (size_t)3 * Hd * Hd * 2;
  unsigned short* whh_bf = (unsigned short*)(ws + off); off += (size_t)3 * Hd * Hd * 2;
  unsigned short* w1_bf  = (unsigned short*)(ws + off); off += (size_t)Hd * Hd * 2;
  unsigned short* w2_bf  = (unsigned short*)(ws + off); off += (size_t)Ed * Hd * 2;
  unsigned short* h0     = (unsigned short*)(ws + off); off += (size_t)Bd * Hd * 2;
  unsigned short* h1     = (unsigned short*)(ws + off); off += (size_t)Bd * Hd * 2;
  unsigned short* a1     = (unsigned short*)(ws + off); off += (size_t)Bd * Hd * 2;
  unsigned*       bar    = (unsigned*)(ws + off);       off += 256;

  cast_f32_to_bf16<<<4096, 256, 0, stream>>>(x, x_bf, Bd * Td * Hd);
  cast_f32_to_bf16<<<2048, 256, 0, stream>>>(w_ih, wih_bf, 3 * Hd * Hd);
  cast_f32_to_bf16<<<2048, 256, 0, stream>>>(w_hh, whh_bf, 3 * Hd * Hd);
  cast_f32_to_bf16<<<1024, 256, 0, stream>>>(w1, w1_bf, Hd * Hd);
  cast_f32_to_bf16<<<1024, 256, 0, stream>>>(w2, w2_bf, Ed * Hd);
  init_barrier<<<1, 1, 0, stream>>>(bar);

  gru_recurrence<<<NWG, 128, 0, stream>>>(x_bf, wih_bf, whh_bf, b_ih, b_hh,
                                          h0, h1, bar);

  // T=256 (even): final h ends in h0 (last write at t=255 targets buf0).
  fc_gemm<1><<<32, 256, 0, stream>>>(h0, w1_bf, b1, a1, nullptr, Hd, Hd);
  fc_gemm<0><<<32, 256, 0, stream>>>(a1, w2_bf, b2, nullptr, out, Hd, Ed);
}
